// RandGR_51788715655934
// MI455X (gfx1250) — matrix operand order repaired and checked by reading, not compiled, not run on hardware
//
#include <hip/hip_runtime.h>

#define HID 128

typedef __attribute__((ext_vector_type(16))) __bf16 v16bf;
typedef __attribute__((ext_vector_type(8)))  float  v8f;

// ---------- helpers ----------
__device__ __forceinline__ __bf16 us_as_bf(unsigned short u) {
  __bf16 b; __builtin_memcpy(&b, &u, 2); return b;
}
__device__ __forceinline__ unsigned short f2bf_us(float f) {
  unsigned u = __float_as_uint(f);
  unsigned r = u + 0x7FFFu + ((u >> 16) & 1u);   // round-to-nearest-even
  return (unsigned short)(r >> 16);
}
__device__ __forceinline__ __bf16 f2bf(float f) { return us_as_bf(f2bf_us(f)); }

// monotonic float<->uint mapping so atomicMax(uint) == float max
__device__ __forceinline__ unsigned fmap(float f) {
  unsigned u = __float_as_uint(f);
  return (u & 0x80000000u) ? ~u : (u | 0x80000000u);
}
__device__ __forceinline__ float funmap(unsigned u) {
  return __uint_as_float((u & 0x80000000u) ? (u & 0x7FFFFFFFu) : ~u);
}

// ---------- pack f32 weight [128,N] into WMMA B-fragment layout (bf16) ----------
// packed[(((tn*4 + kb)*32 + lane)*16 + e] = bf16( W[(kb*32 + (lane>>4)*8 + (e&7) + (e>>3)*16)*N + tn*16 + (lane&15)] )
// => per (tile,kb), each lane's 16 bf16 B elements are 32 contiguous bytes.
__global__ void pack_b_kernel(const float* __restrict__ W, unsigned short* __restrict__ P,
                              int N /* cols */, int total /* = 128*N */) {
  int t = blockIdx.x * blockDim.x + threadIdx.x;
  if (t >= total) return;
  int e    = t & 15;
  int lane = (t >> 4) & 31;
  int kb   = (t >> 9) & 3;
  int tn   = t >> 11;
  int row = kb * 32 + (lane >> 4) * 8 + (e & 7) + (e >> 3) * 16;
  int col = tn * 16 + (lane & 15);
  P[t] = f2bf_us(W[(size_t)row * N + col]);
}

// ---------- GEMM: C[M,N] = A[M,128] (f32) @ B(packed bf16) + bias ----------
// one wave per 16x64 output strip: 4 N-tiles share one A fragment per k-block.
// 16 x v_wmma_f32_16x16x32_bf16 per wave.
__global__ __launch_bounds__(256) void gemm_wmma_bf16(
    const float* __restrict__ A, const unsigned short* __restrict__ B,
    const float* __restrict__ bias, float* __restrict__ C,
    int M, int N, int relu)
{
  const int lane = threadIdx.x & 31;
  const int half = lane >> 4;     // 0: lanes 0-15, 1: lanes 16-31
  const int r    = lane & 15;
  const int tilesN4 = N >> 6;     // strips of 64 columns
  const int tilesM  = M >> 4;
  int wave = blockIdx.x * (blockDim.x >> 5) + (threadIdx.x >> 5);
  if (wave >= tilesM * tilesN4) return;           // wave-uniform exit (EXEC all-1s)
  const int tm  = wave / tilesN4, tn4 = wave % tilesN4;
  const int m0  = tm << 4;

  v8f c0 = {0.f,0.f,0.f,0.f,0.f,0.f,0.f,0.f};
  v8f c1 = c0, c2 = c0, c3 = c0;
  const float* arow = A + (size_t)(m0 + r) * HID;

#pragma unroll
  for (int kb = 0; kb < 4; ++kb) {
    const int kbase = kb * 32;
    // A fragment: 16x32 bf16; lane half selects K runs [kbase+8h..+7], [kbase+16+8h..+7]
    v16bf a;
    {
      const float* p0 = arow + kbase + half * 8;
      const float4 a0 = *(const float4*)(p0);
      const float4 a1 = *(const float4*)(p0 + 4);
      const float4 a2 = *(const float4*)(p0 + 16);
      const float4 a3 = *(const float4*)(p0 + 20);
      a[0]=f2bf(a0.x); a[1]=f2bf(a0.y); a[2]=f2bf(a0.z); a[3]=f2bf(a0.w);
      a[4]=f2bf(a1.x); a[5]=f2bf(a1.y); a[6]=f2bf(a1.z); a[7]=f2bf(a1.w);
      a[8]=f2bf(a2.x); a[9]=f2bf(a2.y); a[10]=f2bf(a2.z); a[11]=f2bf(a2.w);
      a[12]=f2bf(a3.x); a[13]=f2bf(a3.y); a[14]=f2bf(a3.z); a[15]=f2bf(a3.w);
    }
    // B fragments: 32 contiguous bytes per lane per (tile, kb)
    const size_t bstride = 32u * 16u;  // elems per (tile,kb) block
    const unsigned short* bp =
        B + ((size_t)(tn4 * 4) * 4 + kb) * bstride + (size_t)lane * 16;
    const v16bf b0 = *(const v16bf*)(bp);
    const v16bf b1 = *(const v16bf*)(bp + 4 * bstride);
    const v16bf b2 = *(const v16bf*)(bp + 8 * bstride);
    const v16bf b3 = *(const v16bf*)(bp + 12 * bstride);
    c0 = __builtin_amdgcn_wmma_f32_16x16x32_bf16(false, a, false, b0, (short)0, c0, false, false);
    c1 = __builtin_amdgcn_wmma_f32_16x16x32_bf16(false, a, false, b1, (short)0, c1, false, false);
    c2 = __builtin_amdgcn_wmma_f32_16x16x32_bf16(false, a, false, b2, (short)0, c2, false, false);
    c3 = __builtin_amdgcn_wmma_f32_16x16x32_bf16(false, a, false, b3, (short)0, c3, false, false);
  }

  // epilogue: C/D layout: VGPR v -> row m0 + v + 8*half, col n0 + r
  v8f acc[4] = {c0, c1, c2, c3};
#pragma unroll
  for (int t = 0; t < 4; ++t) {
    const int col = (tn4 * 4 + t) * 16 + r;
    const float bv = bias ? bias[col] : 0.f;
#pragma unroll
    for (int v = 0; v < 8; ++v) {
      const int row = m0 + v + half * 8;
      float val = acc[t][v] + bv;
      if (relu) val = fmaxf(val, 0.f);
      C[(size_t)row * N + col] = val;
    }
  }
}

// ---------- small kernels ----------
__global__ void zero_kernel(float* __restrict__ p, int n) {
  int t = blockIdx.x * blockDim.x + threadIdx.x;
  if (t < n) p[t] = 0.f;
}

__global__ void relu_kernel(float* __restrict__ p, int n) {
  int t = blockIdx.x * blockDim.x + threadIdx.x;
  if (t < n) p[t] = fmaxf(p[t], 0.f);
}

__global__ void gather_kernel(const float* __restrict__ emb, const int* __restrict__ idx,
                              float* __restrict__ out, int n /* = rows*HID */) {
  int t = blockIdx.x * blockDim.x + threadIdx.x;
  if (t < n) out[t] = emb[(size_t)idx[t >> 7] * HID + (t & 127)];
}

// out[i] = dot(X[i,:], v) ; one wave per row
__global__ __launch_bounds__(256) void rowdot_kernel(const float* __restrict__ X,
                                                     const float* __restrict__ v,
                                                     float* __restrict__ out, int R) {
  int wave = blockIdx.x * (blockDim.x >> 5) + (threadIdx.x >> 5);
  int lane = threadIdx.x & 31;
  if (wave >= R) return;
  const float4 x = *(const float4*)(X + (size_t)wave * HID + lane * 4);
  const float4 w = *(const float4*)(v + lane * 4);
  float s = x.x * w.x + x.y * w.y + x.z * w.z + x.w * w.w;
  for (int off = 16; off; off >>= 1) s += __shfl_xor(s, off, 32);
  if (lane == 0) out[wave] = s;
}

// adv[i] = sum_o Wd[i,o] * ad[o]  (128x128 matvec, tiny)
__global__ void matvec128_kernel(const float* __restrict__ Wd, const float* __restrict__ ad,
                                 float* __restrict__ out) {
  int i = threadIdx.x;
  float s = 0.f;
  for (int o = 0; o < HID; ++o) s += Wd[i * HID + o] * ad[o];
  out[i] = s;
}

// combined per-node-type bias: lin.b + sum of conv.b for edge types targeting that type
__global__ void bias_comb_kernel(const float* lg, const float* lu, const float* li,
                                 const float* b_u2g, const float* b_g2u,
                                 const float* b_i2u, const float* b_u2i,
                                 float* cg, float* cu, float* ci) {
  int i = threadIdx.x;
  cg[i] = lg[i] + b_u2g[i];
  cu[i] = lu[i] + b_g2u[i] + b_i2u[i];
  ci[i] = li[i] + b_u2i[i];
}

__global__ void seg_init_kernel(unsigned* __restrict__ m, float* __restrict__ s, int n) {
  int t = blockIdx.x * blockDim.x + threadIdx.x;
  if (t < n) { m[t] = 0x007FFFFFu /* fmap(-inf) */; s[t] = 0.f; }
}

__global__ void edge_max_kernel(const int* __restrict__ src, const int* __restrict__ dst,
                                const float* __restrict__ a_s, const float* __restrict__ a_d,
                                float* __restrict__ ebuf, unsigned* __restrict__ mmax, int E) {
  int e = blockIdx.x * blockDim.x + threadIdx.x;
  if (e >= E) return;
  float v = a_s[src[e]] + a_d[dst[e]];
  v = (v > 0.f) ? v : 0.2f * v;          // leaky_relu(0.2)
  ebuf[e] = v;
  atomicMax(&mmax[dst[e]], fmap(v));
}

__global__ void edge_expsum_kernel(const int* __restrict__ dst, const unsigned* __restrict__ mmax,
                                   float* __restrict__ ebuf, float* __restrict__ ssum, int E) {
  int e = blockIdx.x * blockDim.x + threadIdx.x;
  if (e >= E) return;
  int d = dst[e];
  float ex = __expf(ebuf[e] - funmap(mmax[d]));
  ebuf[e] = ex;
  atomicAdd(&ssum[d], ex);
}

// out[dst] += hs[src] * alpha ; one wave per edge, float4 per lane
__global__ __launch_bounds__(256) void edge_scatter_kernel(
    const int* __restrict__ src, const int* __restrict__ dst,
    const float* __restrict__ ebuf, const float* __restrict__ ssum,
    const float* __restrict__ hs, float* __restrict__ out, int E) {
  int t = blockIdx.x * blockDim.x + threadIdx.x;
  int e = t >> 5, lane = t & 31;
  if (e >= E) return;
  int s = src[e], d = dst[e];
  float alpha = ebuf[e] / (ssum[d] + 1e-16f);
  const float4 hv = *(const float4*)(hs + (size_t)s * HID + lane * 4);
  float* op = out + (size_t)d * HID + lane * 4;
  atomicAdd(op + 0, hv.x * alpha);
  atomicAdd(op + 1, hv.y * alpha);
  atomicAdd(op + 2, hv.z * alpha);
  atomicAdd(op + 3, hv.w * alpha);
}

// ---------- host ----------
extern "C" void kernel_launch(void* const* d_in, const int* in_sizes, int n_in,
                              void* d_out, int out_size, void* d_ws, size_t ws_size,
                              hipStream_t stream) {
  auto F = [&](int i) { return (const float*)d_in[i]; };
  auto I = [&](int i) { return (const int*)d_in[i]; };

  const int nG = in_sizes[0], nU = in_sizes[1], nI = in_sizes[2];
  const int E_u2g = in_sizes[3], E_g2u = in_sizes[5], E_u2i = in_sizes[7], E_i2u = in_sizes[9];

  // param flatten order (jax pytree: dict keys sorted): conv1,conv2,emb,lin1,lin2,pred
  // conv leaves: Wd, Ws, ad, as, b ; edge types sorted: g2u,i2u,u2g,u2i
  const int CBASE[2] = {11, 31};
  const int EMB_G = 51, EMB_I = 52, EMB_U = 53;
  const int LBASE[2] = {54, 60};  // per layer: gW,gb,iW,ib,uW,ub
  const int PRED_W = 66, PRED_B = 67;
  const int predN = in_sizes[PRED_B];          // 32000
  const int predWsz = in_sizes[PRED_W];        // 128*32000
  (void)EMB_G;

  // workspace carve-up
  size_t off = 0; char* wb = (char*)d_ws;
  auto alloc = [&](size_t bytes) -> void* {
    off = (off + 255) & ~(size_t)255;
    void* p = wb + off; off += bytes; return p;
  };
  float* hG = (float*)alloc((size_t)nG * HID * 4);
  float* hU = (float*)alloc((size_t)nU * HID * 4);
  float* hI = (float*)alloc((size_t)nI * HID * 4);
  float* oG = (float*)alloc((size_t)nG * HID * 4);
  float* oU = (float*)alloc((size_t)nU * HID * 4);
  float* oI = (float*)alloc((size_t)nI * HID * 4);
  float* hs = (float*)alloc((size_t)nU * HID * 4);   // max Ns = users
  float* a_s  = (float*)alloc((size_t)nU * 4);
  float* a_d  = (float*)alloc((size_t)nU * 4);
  float* adv  = (float*)alloc(HID * 4);
  unsigned* m_u = (unsigned*)alloc((size_t)nU * 4);
  float* ssum = (float*)alloc((size_t)nU * 4);
  int Emax = E_u2i > E_i2u ? E_u2i : E_i2u;
  if (E_u2g > Emax) Emax = E_u2g;
  if (E_g2u > Emax) Emax = E_g2u;
  float* ebuf = (float*)alloc((size_t)Emax * 4);
  float* cbG = (float*)alloc(HID * 4);
  float* cbU = (float*)alloc(HID * 4);
  float* cbI = (float*)alloc(HID * 4);
  unsigned short* WsBf[2][4];
  for (int l = 0; l < 2; ++l)
    for (int e = 0; e < 4; ++e) WsBf[l][e] = (unsigned short*)alloc(HID * HID * 2);
  unsigned short* LinBf[2][3];
  for (int l = 0; l < 2; ++l)
    for (int t = 0; t < 3; ++t) LinBf[l][t] = (unsigned short*)alloc(HID * HID * 2);
  unsigned short* PredBf = (unsigned short*)alloc((size_t)predWsz * 2);

  const int B = 256;
  auto cdiv = [](long long a, long long b) { return (int)((a + b - 1) / b); };

  auto pack = [&](const float* src, unsigned short* dst, int N) {
    int total = HID * N;
    pack_b_kernel<<<cdiv(total, B), B, 0, stream>>>(src, dst, N, total);
  };
  auto gemm = [&](const float* A, const unsigned short* Bw, const float* bias,
                  float* C, int M, int N, int relu) {
    int tiles = (M >> 4) * (N >> 6);
    gemm_wmma_bf16<<<cdiv(tiles, 8), 256, 0, stream>>>(A, Bw, bias, C, M, N, relu);
  };

  // ---- pre-pack all weights used as GEMM B operands into fragment layout ----
  for (int l = 0; l < 2; ++l)
    for (int e = 0; e < 4; ++e)
      pack(F(CBASE[l] + 5 * e + 1), WsBf[l][e], HID);               // Ws
  for (int l = 0; l < 2; ++l)
    for (int t = 0; t < 3; ++t)
      pack(F(LBASE[l] + 2 * t), LinBf[l][t], HID);                  // lin W (g,i,u)
  pack(F(PRED_W), PredBf, predN);

  // ---- init node features ----
  zero_kernel<<<cdiv((long long)nG * HID, B), B, 0, stream>>>(hG, nG * HID);
  gather_kernel<<<cdiv((long long)nU * HID, B), B, 0, stream>>>(F(EMB_U), I(1), hU, nU * HID);
  gather_kernel<<<cdiv((long long)nI * HID, B), B, 0, stream>>>(F(EMB_I), I(2), hI, nI * HID);

  // edge-type tables (param order: g2u, i2u, u2g, u2i)
  const int* eSrc[4] = {I(5), I(9), I(3), I(7)};
  const int* eDst[4] = {I(6), I(10), I(4), I(8)};
  const int  eCnt[4] = {E_g2u, E_i2u, E_u2g, E_u2i};
  const int  stType[4] = {0, 2, 1, 1};  // 0=group 1=user 2=item (source)
  const int  dtType[4] = {1, 1, 0, 2};  // (dest)

  for (int l = 0; l < 2; ++l) {
    float* H[3] = {hG, hU, hI};
    float* O[3] = {oG, oU, oI};
    const int NN[3] = {nG, nU, nI};

    // lin bias + conv biases targeting each node type
    bias_comb_kernel<<<1, HID, 0, stream>>>(
        F(LBASE[l] + 1), F(LBASE[l] + 5), F(LBASE[l] + 3),
        F(CBASE[l] + 5 * 2 + 4),  // u2g.b
        F(CBASE[l] + 5 * 0 + 4),  // g2u.b
        F(CBASE[l] + 5 * 1 + 4),  // i2u.b
        F(CBASE[l] + 5 * 3 + 4),  // u2i.b
        cbG, cbU, cbI);
    float* CB[3] = {cbG, cbU, cbI};

    // out[nt] = h[nt] @ linW + combined bias
    for (int t = 0; t < 3; ++t) {
      const int ord[3] = {0, 2, 1};  // LinBf order is (group,item,user) -> nt(g,u,i)
      gemm(H[t], LinBf[l][ord[t]], CB[t], O[t], NN[t], HID, 0);
    }

    for (int e = 0; e < 4; ++e) {
      const int st = stType[e], dt = dtType[e];
      const int Ns = NN[st], Nd = NN[dt], E = eCnt[e];
      const float* Wd  = F(CBASE[l] + 5 * e + 0);
      const float* adV = F(CBASE[l] + 5 * e + 2);
      const float* asV = F(CBASE[l] + 5 * e + 3);

      gemm(H[st], WsBf[l][e], nullptr, hs, Ns, HID, 0);                       // hs = xs@Ws
      rowdot_kernel<<<cdiv(Ns, 8), 256, 0, stream>>>(hs, asV, a_s, Ns);       // a_s = hs@as
      matvec128_kernel<<<1, HID, 0, stream>>>(Wd, adV, adv);                  // adv = Wd@ad
      rowdot_kernel<<<cdiv(Nd, 8), 256, 0, stream>>>(H[dt], adv, a_d, Nd);    // a_d = xd@adv
      seg_init_kernel<<<cdiv(Nd, B), B, 0, stream>>>(m_u, ssum, Nd);
      edge_max_kernel<<<cdiv(E, B), B, 0, stream>>>(eSrc[e], eDst[e], a_s, a_d, ebuf, m_u, E);
      edge_expsum_kernel<<<cdiv(E, B), B, 0, stream>>>(eDst[e], m_u, ebuf, ssum, E);
      edge_scatter_kernel<<<cdiv((long long)E * 32, B), B, 0, stream>>>(
          eSrc[e], eDst[e], ebuf, ssum, hs, O[dt], E);
    }

    relu_kernel<<<cdiv((long long)nG * HID, B), B, 0, stream>>>(oG, nG * HID);
    relu_kernel<<<cdiv((long long)nU * HID, B), B, 0, stream>>>(oU, nU * HID);
    relu_kernel<<<cdiv((long long)nI * HID, B), B, 0, stream>>>(oI, nI * HID);

    // swap h <-> o for next layer
    float* t;
    t = hG; hG = oG; oG = t;
    t = hU; hU = oU; oU = t;
    t = hI; hI = oI; oI = t;
  }

  // predictor: d_out[4000, 32000] = h_group @ pred.W + pred.b
  gemm(hG, PredBf, F(PRED_B), (float*)d_out, nG, predN, 0);
}
